// TransformerBlock_50878182588505
// MI455X (gfx1250) — matrix operand order repaired and checked by reading, not compiled, not run on hardware
//
#include <hip/hip_runtime.h>
#include <hip/hip_bf16.h>
#include <stdint.h>

// ---- problem constants (from reference) ----
#define BB 2
#define SS 2048
#define DIM 768
#define FF 3072
#define HH 12
#define HD 64
#define BSROWS (BB * SS) // 4096

typedef __bf16 bf16;
typedef __attribute__((ext_vector_type(16))) __bf16 v16bf;
typedef __attribute__((ext_vector_type(8))) float v8f;
typedef __attribute__((ext_vector_type(4))) unsigned int u32x4;

union Frag {
  v16bf v;
  struct { u32x4 lo, hi; } q;
  __bf16 e[16];
};

// A-matrix fragment, 16x32 bf16 (ISA 7.12.2 layout)
__device__ __forceinline__ Frag load_a_frag(const bf16* t, int ld, int half, int l16) {
  Frag f;
  const bf16* p = t + (size_t)l16 * ld + 8 * half;
  f.q.lo = *(const u32x4*)(p);
  f.q.hi = *(const u32x4*)(p + 16);
  return f;
}

// B-matrix fragment, 32x16 bf16 from an [N][K] row-major matrix in GLOBAL memory
__device__ __forceinline__ Frag load_b_frag(const bf16* t, int ld, int half, int l16) {
  Frag f;
  const bf16* p = t + (size_t)l16 * ld + 8 * half;
  f.q.lo = *(const u32x4*)(p);
  f.q.hi = *(const u32x4*)(p + 16);
  return f;
}

// B-matrix fragment from an LDS-staged 64x32 tile with XOR-swizzled 16B chunks.
// tile row r (64 B) holds 4 chunks; chunk c stored at slot c ^ ((r>>2)&3).
__device__ __forceinline__ Frag load_b_lds(const bf16* lt, int nt, int half, int l16) {
  const int row = nt * 16 + l16;
  const int swz = (l16 >> 2) & 3;
  Frag f;
  const bf16* p = lt + row * 32;
  f.q.lo = *(const u32x4*)(p + (((half) ^ swz) << 3));
  f.q.hi = *(const u32x4*)(p + (((half + 2) ^ swz) << 3));
  return f;
}

__device__ __forceinline__ v8f wmma_bf16(Frag a, Frag b, v8f c) {
  return __builtin_amdgcn_wmma_f32_16x16x32_bf16(false, a.v, false, b.v,
                                                 (short)0, c, false, false);
}

// Async-copy one 16B chunk of a 64x32 W tile into LDS (CDNA5 ASYNCcnt path).
// Thread tid (0..255): row = tid>>2, chunk = tid&3, XOR-swizzled slot.
__device__ __forceinline__ void stage_tile_async(bf16* ltile, const bf16* W, int ldw,
                                                 int n0, int kb, int tid) {
  const int row = tid >> 2, part = tid & 3;
  const int slot = part ^ ((row >> 2) & 3);
  const bf16* gp = W + (size_t)(n0 + row) * ldw + kb + part * 8;
  unsigned lds_addr = (unsigned)(size_t)(ltile + row * 32 + slot * 8);
  asm volatile("global_load_async_to_lds_b128 %0, %1, off"
               :: "v"(lds_addr), "v"(gp) : "memory");
}

__device__ __forceinline__ void wait_async_le(int) {}
__device__ __forceinline__ void wait_async0() {
  asm volatile("s_wait_asynccnt 0x0" ::: "memory");
}
__device__ __forceinline__ void wait_async1() {
  asm volatile("s_wait_asynccnt 0x1" ::: "memory");
}
__device__ __forceinline__ void wait_async2() {
  asm volatile("s_wait_asynccnt 0x2" ::: "memory");
}

// ---- elementwise: f32 -> bf16 ----
__global__ void k_f32_to_bf16(const float* __restrict__ src, bf16* __restrict__ dst, int n) {
  int i = blockIdx.x * blockDim.x + threadIdx.x;
  if (i < n) dst[i] = (bf16)src[i];
}

// ---- RMSNorm: f32 row -> bf16 row ----
__global__ __launch_bounds__(256) void k_rmsnorm(const float* __restrict__ x,
                                                 const float* __restrict__ w,
                                                 bf16* __restrict__ out, int D) {
  __shared__ float red[256];
  const int row = blockIdx.x;
  const float* xr = x + (size_t)row * D;
  float ss = 0.0f;
  for (int i = threadIdx.x; i < D; i += 256) { float v = xr[i]; ss += v * v; }
  red[threadIdx.x] = ss;
  __syncthreads();
  for (int s = 128; s > 0; s >>= 1) {
    if (threadIdx.x < s) red[threadIdx.x] += red[threadIdx.x + s];
    __syncthreads();
  }
  const float rs = rsqrtf(red[0] / (float)D + 1e-5f);
  for (int i = threadIdx.x; i < D; i += 256)
    out[(size_t)row * D + i] = (bf16)(xr[i] * rs * w[i]);
}

// ---- GEMM: C = A @ W^T, LDS-staged double-buffered B, bf16 out ----
__global__ __launch_bounds__(256) void k_gemm_bf16(const bf16* __restrict__ A,
                                                   const bf16* __restrict__ W,
                                                   bf16* __restrict__ C,
                                                   int M, int N, int K) {
  __shared__ bf16 tb[2][64 * 32];
  const int tid = threadIdx.x;
  const int wid = tid >> 5, lane = tid & 31;
  const int half = lane >> 4, l16 = lane & 15;
  const int m0 = (blockIdx.x * 8 + wid) * 16;
  const int n0 = blockIdx.y * 64;
  const bf16* Arow = A + (size_t)m0 * K;
  v8f acc[4] = {};
  const int nsteps = K / 32;
  stage_tile_async(tb[0], W, K, n0, 0, tid);
  for (int i = 0; i < nsteps; ++i) {
    const int kb = i * 32;
    const bool more = (i + 1 < nsteps);
    if (more) {
      stage_tile_async(tb[(i + 1) & 1], W, K, n0, kb + 32, tid);
      wait_async1();            // async loads complete in order: tile i is done
    } else {
      wait_async0();
    }
    __syncthreads();            // tile i visible to all waves
    Frag a = load_a_frag(Arow + kb, K, half, l16);
    __builtin_prefetch(Arow + (size_t)l16 * K + kb + 32, 0, 1);
    const bf16* lt = tb[i & 1];
#pragma unroll
    for (int nt = 0; nt < 4; ++nt) {
      Frag b = load_b_lds(lt, nt, half, l16);
      acc[nt] = wmma_bf16(a, b, acc[nt]);
    }
    __syncthreads();            // all waves done with tile i before it is re-staged
  }
#pragma unroll
  for (int nt = 0; nt < 4; ++nt)
#pragma unroll
    for (int r = 0; r < 8; ++r)
      C[(size_t)(m0 + r + 8 * half) * N + n0 + nt * 16 + l16] = (bf16)acc[nt][r];
}

// ---- GEMM + residual: out_f32 = res + A @ W^T ----
__global__ __launch_bounds__(256) void k_gemm_res_f32(const bf16* __restrict__ A,
                                                      const bf16* __restrict__ W,
                                                      const float* __restrict__ res,
                                                      float* __restrict__ out,
                                                      int M, int N, int K) {
  __shared__ bf16 tb[2][64 * 32];
  const int tid = threadIdx.x;
  const int wid = tid >> 5, lane = tid & 31;
  const int half = lane >> 4, l16 = lane & 15;
  const int m0 = (blockIdx.x * 8 + wid) * 16;
  const int n0 = blockIdx.y * 64;
  const bf16* Arow = A + (size_t)m0 * K;
  v8f acc[4] = {};
  const int nsteps = K / 32;
  stage_tile_async(tb[0], W, K, n0, 0, tid);
  for (int i = 0; i < nsteps; ++i) {
    const int kb = i * 32;
    const bool more = (i + 1 < nsteps);
    if (more) {
      stage_tile_async(tb[(i + 1) & 1], W, K, n0, kb + 32, tid);
      wait_async1();
    } else {
      wait_async0();
    }
    __syncthreads();
    Frag a = load_a_frag(Arow + kb, K, half, l16);
    __builtin_prefetch(Arow + (size_t)l16 * K + kb + 32, 0, 1);
    const bf16* lt = tb[i & 1];
#pragma unroll
    for (int nt = 0; nt < 4; ++nt) {
      Frag b = load_b_lds(lt, nt, half, l16);
      acc[nt] = wmma_bf16(a, b, acc[nt]);
    }
    __syncthreads();
  }
#pragma unroll
  for (int nt = 0; nt < 4; ++nt)
#pragma unroll
    for (int r = 0; r < 8; ++r) {
      size_t idx = (size_t)(m0 + r + 8 * half) * N + n0 + nt * 16 + l16;
      out[idx] = res[idx] + acc[nt][r];
    }
}

// ---- fused SwiGLU GEMM: out = silu(A@W1^T) * (A@W3^T), bf16 out ----
__global__ __launch_bounds__(256) void k_gemm_dual_silu(const bf16* __restrict__ A,
                                                        const bf16* __restrict__ W1,
                                                        const bf16* __restrict__ W3,
                                                        bf16* __restrict__ C,
                                                        int M, int N, int K) {
  __shared__ bf16 tb1[2][64 * 32];
  __shared__ bf16 tb3[2][64 * 32];
  const int tid = threadIdx.x;
  const int wid = tid >> 5, lane = tid & 31;
  const int half = lane >> 4, l16 = lane & 15;
  const int m0 = (blockIdx.x * 8 + wid) * 16;
  const int n0 = blockIdx.y * 64;
  const bf16* Arow = A + (size_t)m0 * K;
  v8f acc1[4] = {};
  v8f acc3[4] = {};
  const int nsteps = K / 32;
  stage_tile_async(tb1[0], W1, K, n0, 0, tid);
  stage_tile_async(tb3[0], W3, K, n0, 0, tid);
  for (int i = 0; i < nsteps; ++i) {
    const int kb = i * 32;
    const bool more = (i + 1 < nsteps);
    if (more) {
      stage_tile_async(tb1[(i + 1) & 1], W1, K, n0, kb + 32, tid);
      stage_tile_async(tb3[(i + 1) & 1], W3, K, n0, kb + 32, tid);
      wait_async2();            // 2 in flight for next tile; current pair done
    } else {
      wait_async0();
    }
    __syncthreads();
    Frag a = load_a_frag(Arow + kb, K, half, l16);
    const bf16* lt1 = tb1[i & 1];
    const bf16* lt3 = tb3[i & 1];
#pragma unroll
    for (int nt = 0; nt < 4; ++nt) {
      Frag b1 = load_b_lds(lt1, nt, half, l16);
      acc1[nt] = wmma_bf16(a, b1, acc1[nt]);
      Frag b3 = load_b_lds(lt3, nt, half, l16);
      acc3[nt] = wmma_bf16(a, b3, acc3[nt]);
    }
    __syncthreads();
  }
#pragma unroll
  for (int nt = 0; nt < 4; ++nt)
#pragma unroll
    for (int r = 0; r < 8; ++r) {
      float u = acc1[nt][r];
      float g3 = acc3[nt][r];
      float val = (u / (1.0f + __expf(-u))) * g3;
      C[(size_t)(m0 + r + 8 * half) * N + n0 + nt * 16 + l16] = (bf16)val;
    }
}

// ---- RoPE + reshape: [b*s][D] bf16 -> [b][h][s][hd] bf16 ----
__global__ void k_rope_reshape(const bf16* __restrict__ src, bf16* __restrict__ dst, int n) {
  int idx = blockIdx.x * blockDim.x + threadIdx.x;
  if (idx >= n) return;
  int row = idx / (DIM / 2);
  int i = idx % (DIM / 2);
  int b = row / SS, s = row % SS;
  int head = i / (HD / 2);
  int dp = i & (HD / 2 - 1);
  float fr = __powf(10000.0f, -2.0f * (float)dp / (float)HD);
  float ang = (float)s * fr;
  float c = __cosf(ang), sn = __sinf(ang);
  float xe = (float)src[(size_t)row * DIM + 2 * i];
  float xo = (float)src[(size_t)row * DIM + 2 * i + 1];
  size_t base = (((size_t)b * HH + head) * SS + s) * HD + 2 * dp;
  dst[base] = (bf16)(xe * c - xo * sn);
  dst[base + 1] = (bf16)(xe * sn + xo * c);
}

// ---- V transpose: [b*s][D] -> [b][h][hd][s] ----
__global__ void k_v_transpose(const bf16* __restrict__ src, bf16* __restrict__ dst, int n) {
  int idx = blockIdx.x * blockDim.x + threadIdx.x;
  if (idx >= n) return;
  int row = idx / DIM, c = idx % DIM;
  int b = row / SS, s = row % SS;
  int head = c / HD, d = c % HD;
  dst[(((size_t)b * HH + head) * HD + d) * SS + s] = src[idx];
}

// ---- flash attention: one wave per 16-query tile, online softmax in LDS ----
__global__ __launch_bounds__(128) void k_attention(const bf16* __restrict__ Q,
                                                   const bf16* __restrict__ Kmat,
                                                   const bf16* __restrict__ Vt,
                                                   bf16* __restrict__ ctx) {
  __shared__ float sS[4][2][16][16];
  __shared__ float sM[4][16], sL[4][16], sF[4][16];
  const int w = threadIdx.x >> 5, lane = threadIdx.x & 31;
  const int half = lane >> 4, l16 = lane & 15;
  const int bh = blockIdx.x;          // b*H + h
  const int t = blockIdx.y * 4 + w;   // q-tile 0..127
  const int q0 = t * 16;
  const int b = bh / HH, h = bh % HH;
  const bf16* qT = Q + ((size_t)bh * SS + q0) * HD;
  const bf16* kB = Kmat + (size_t)bh * SS * HD;
  const bf16* vB = Vt + (size_t)bh * HD * SS;
  const float scale = 0.125f; // 1/sqrt(64)

  Frag aq0 = load_a_frag(qT, HD, half, l16);
  Frag aq1 = load_a_frag(qT + 32, HD, half, l16);
  v8f acc[4] = {};
  if (!half) { sM[w][l16] = -3.0e38f; sL[w][l16] = 0.0f; }
  asm volatile("s_wait_dscnt 0" ::: "memory");

  for (int k0 = 0; k0 <= q0; k0 += 32) {
#pragma unroll
    for (int nt = 0; nt < 2; ++nt) {
      const int kn0 = k0 + nt * 16;
      Frag bk0 = load_b_frag(kB + (size_t)kn0 * HD, HD, half, l16);
      Frag bk1 = load_b_frag(kB + (size_t)kn0 * HD + 32, HD, half, l16);
      v8f sc = {};
      sc = wmma_bf16(aq0, bk0, sc);
      sc = wmma_bf16(aq1, bk1, sc);
#pragma unroll
      for (int r = 0; r < 8; ++r) {
        int row = r + 8 * half;
        int key = kn0 + l16;
        float sv = sc[r] * scale;
        if (key > q0 + row) sv = -3.0e38f; // causal mask
        sS[w][nt][row][l16] = sv;
      }
    }
    asm volatile("s_wait_dscnt 0" ::: "memory");
    if (!half) {
      float mo = sM[w][l16];
      float mx = mo;
#pragma unroll
      for (int j = 0; j < 32; ++j) mx = fmaxf(mx, sS[w][j >> 4][l16][j & 15]);
      float fac = __expf(mo - mx);
      float sum = 0.0f;
#pragma unroll
      for (int j = 0; j < 32; ++j) {
        float p = __expf(sS[w][j >> 4][l16][j & 15] - mx);
        sS[w][j >> 4][l16][j & 15] = p;
        sum += p;
      }
      sM[w][l16] = mx;
      sL[w][l16] = sL[w][l16] * fac + sum;
      sF[w][l16] = fac;
    }
    asm volatile("s_wait_dscnt 0" ::: "memory");
    Frag pf;
#pragma unroll
    for (int j = 0; j < 8; ++j) {
      pf.e[j] = (__bf16)sS[w][0][l16][8 * half + j];
      pf.e[8 + j] = (__bf16)sS[w][1][l16][8 * half + j];
    }
    float fac_r[8];
#pragma unroll
    for (int r = 0; r < 8; ++r) fac_r[r] = sF[w][r + 8 * half];
#pragma unroll
    for (int nt = 0; nt < 4; ++nt) {
#pragma unroll
      for (int r = 0; r < 8; ++r) acc[nt][r] *= fac_r[r];
      Frag bv = load_b_frag(vB + (size_t)(nt * 16) * SS + k0, SS, half, l16);
      acc[nt] = wmma_bf16(pf, bv, acc[nt]);
    }
  }
  asm volatile("s_wait_dscnt 0" ::: "memory");
  float l_r[8];
#pragma unroll
  for (int r = 0; r < 8; ++r) l_r[r] = sL[w][r + 8 * half];
#pragma unroll
  for (int nt = 0; nt < 4; ++nt)
#pragma unroll
    for (int r = 0; r < 8; ++r) {
      int row = r + 8 * half;
      float o = acc[nt][r] / l_r[r];
      ctx[((size_t)b * SS + q0 + row) * DIM + (size_t)h * HD + nt * 16 + l16] = (bf16)o;
    }
}

extern "C" void kernel_launch(void* const* d_in, const int* in_sizes, int n_in,
                              void* d_out, int out_size, void* d_ws, size_t ws_size,
                              hipStream_t stream) {
  const float* x = (const float*)d_in[0];
  const float* w_q = (const float*)d_in[1];
  const float* w_k = (const float*)d_in[2];
  const float* w_v = (const float*)d_in[3];
  const float* w_o = (const float*)d_in[4];
  const float* w_ln1 = (const float*)d_in[5];
  const float* w_ln2 = (const float*)d_in[6];
  const float* w_ffn1 = (const float*)d_in[7];
  const float* w_ffn2 = (const float*)d_in[8];
  const float* w_ffn3 = (const float*)d_in[9];
  float* out = (float*)d_out;

  char* ws = (char*)d_ws;
  size_t off = 0;
  auto alloc = [&](size_t bytes) -> char* {
    char* p = ws + off;
    off += (bytes + 255) & ~(size_t)255;
    return p;
  };
  const size_t DDb = (size_t)DIM * DIM * 2;
  const size_t FDb = (size_t)FF * DIM * 2;
  const size_t BSDb = (size_t)BSROWS * DIM * 2;
  const size_t BSFb = (size_t)BSROWS * FF * 2;

  bf16* wq_b = (bf16*)alloc(DDb);
  bf16* wk_b = (bf16*)alloc(DDb);
  bf16* wv_b = (bf16*)alloc(DDb);
  bf16* wo_b = (bf16*)alloc(DDb);
  bf16* w1_b = (bf16*)alloc(FDb);
  bf16* w3_b = (bf16*)alloc(FDb);
  bf16* w2_b = (bf16*)alloc(FDb);
  bf16* hn = (bf16*)alloc(BSDb);   // reused for g after 2nd rmsnorm
  bf16* qf = (bf16*)alloc(BSDb);   // [b][h][s][hd]
  bf16* kf = (bf16*)alloc(BSDb);   // [b][h][s][hd]
  bf16* vT = (bf16*)alloc(BSDb);   // [b][h][hd][s]
  bf16* q_raw = (bf16*)alloc(BSDb);
  bf16* k_raw = (bf16*)alloc(BSDb);
  bf16* v_raw = (bf16*)alloc(BSDb);
  bf16* u_fused = (bf16*)alloc(BSFb);
  bf16* ctx = q_raw;               // q_raw dead after rope
  float* hres = (float*)k_raw;     // k_raw+v_raw dead after rope/transpose

  const int DD = DIM * DIM, FD = FF * DIM;
  k_f32_to_bf16<<<(DD + 255) / 256, 256, 0, stream>>>(w_q, wq_b, DD);
  k_f32_to_bf16<<<(DD + 255) / 256, 256, 0, stream>>>(w_k, wk_b, DD);
  k_f32_to_bf16<<<(DD + 255) / 256, 256, 0, stream>>>(w_v, wv_b, DD);
  k_f32_to_bf16<<<(DD + 255) / 256, 256, 0, stream>>>(w_o, wo_b, DD);
  k_f32_to_bf16<<<(FD + 255) / 256, 256, 0, stream>>>(w_ffn1, w1_b, FD);
  k_f32_to_bf16<<<(FD + 255) / 256, 256, 0, stream>>>(w_ffn3, w3_b, FD);
  k_f32_to_bf16<<<(FD + 255) / 256, 256, 0, stream>>>(w_ffn2, w2_b, FD);

  k_rmsnorm<<<BSROWS, 256, 0, stream>>>(x, w_ln1, hn, DIM);

  dim3 gP(BSROWS / 128, DIM / 64);
  k_gemm_bf16<<<gP, 256, 0, stream>>>(hn, wq_b, q_raw, BSROWS, DIM, DIM);
  k_gemm_bf16<<<gP, 256, 0, stream>>>(hn, wk_b, k_raw, BSROWS, DIM, DIM);
  k_gemm_bf16<<<gP, 256, 0, stream>>>(hn, wv_b, v_raw, BSROWS, DIM, DIM);

  const int nrope = BSROWS * DIM / 2;
  k_rope_reshape<<<(nrope + 255) / 256, 256, 0, stream>>>(q_raw, qf, nrope);
  k_rope_reshape<<<(nrope + 255) / 256, 256, 0, stream>>>(k_raw, kf, nrope);
  const int nv = BSROWS * DIM;
  k_v_transpose<<<(nv + 255) / 256, 256, 0, stream>>>(v_raw, vT, nv);

  dim3 gA(BB * HH, SS / 16 / 4);
  k_attention<<<gA, 128, 0, stream>>>(qf, kf, vT, ctx);

  k_gemm_res_f32<<<gP, 256, 0, stream>>>(ctx, wo_b, x, hres, BSROWS, DIM, DIM);

  k_rmsnorm<<<BSROWS, 256, 0, stream>>>(hres, w_ln2, hn, DIM);

  dim3 gF(BSROWS / 128, FF / 64);
  k_gemm_dual_silu<<<gF, 256, 0, stream>>>(hn, w1_b, w3_b, u_fused, BSROWS, FF, DIM);

  k_gemm_res_f32<<<gP, 256, 0, stream>>>(u_fused, w2_b, hres, out, BSROWS, DIM, FF);
}
